// Agcn_87058987090108
// MI455X (gfx1250) — matrix operand order repaired and checked by reading, not compiled, not run on hardware
//
#include <hip/hip_runtime.h>
#include <hip/hip_bf16.h>

// Problem constants (from reference)
#define BB     32
#define CIN    32
#define VV     512
#define LL     64
#define EDIM   10
#define COUT   64
#define CCAT   128

typedef __attribute__((ext_vector_type(16))) __bf16 v16bf;
typedef __attribute__((ext_vector_type(8)))  __bf16 v8bf;
typedef __attribute__((ext_vector_type(8)))  float  v8f;

// ---------------------------------------------------------------------------
// Kernel 1: adjacency matrices.
// For hop h in {0,1,2}: propagate nodevec1/nodevec2 through w_trans chain,
// score = relu(n1_h @ n2_h), row-softmax (axis=1, over w), store adp^T in bf16:
//   adpT[h][w][v] = adp_h[v][w]   (WMMA A operand layout: M=w, K=v)
// One block per (hop, v); 256 threads, each handles 2 columns w.
// ---------------------------------------------------------------------------
__global__ __launch_bounds__(256) void adj_kernel(
    const float* __restrict__ nodevec1,   // [V, E]
    const float* __restrict__ nodevec2,   // [E, V]
    const float* __restrict__ w_trans,    // [2, E, E]
    const float* __restrict__ b_trans,    // [2, E]
    __bf16* __restrict__ adpT)            // [3, V, V]
{
    const int hop = blockIdx.x / VV;
    const int v   = blockIdx.x % VV;
    const int tid = threadIdx.x;

    float e1[EDIM];
    #pragma unroll
    for (int e = 0; e < EDIM; ++e) e1[e] = nodevec1[v * EDIM + e];
    for (int h = 0; h < hop; ++h) {
        float t[EDIM];
        #pragma unroll
        for (int e = 0; e < EDIM; ++e) {
            float s = b_trans[h * EDIM + e];
            #pragma unroll
            for (int f = 0; f < EDIM; ++f)
                s += e1[f] * w_trans[(h * EDIM + f) * EDIM + e];
            t[e] = s;
        }
        #pragma unroll
        for (int e = 0; e < EDIM; ++e) e1[e] = t[e];
    }

    float sc[2];
    const int wids[2] = { tid, tid + 256 };
    for (int q = 0; q < 2; ++q) {
        const int w = wids[q];
        float e2[EDIM];
        #pragma unroll
        for (int e = 0; e < EDIM; ++e) e2[e] = nodevec2[e * VV + w];
        for (int h = 0; h < hop; ++h) {
            float t[EDIM];
            #pragma unroll
            for (int e = 0; e < EDIM; ++e) {
                float s = b_trans[h * EDIM + e];
                #pragma unroll
                for (int f = 0; f < EDIM; ++f)
                    s += e2[f] * w_trans[(h * EDIM + f) * EDIM + e];
                t[e] = s;
            }
            #pragma unroll
            for (int e = 0; e < EDIM; ++e) e2[e] = t[e];
        }
        float s = 0.f;
        #pragma unroll
        for (int e = 0; e < EDIM; ++e) s += e1[e] * e2[e];
        sc[q] = fmaxf(s, 0.f);   // relu before softmax
    }

    __shared__ float red[256];
    red[tid] = fmaxf(sc[0], sc[1]);
    __syncthreads();
    for (int off = 128; off > 0; off >>= 1) {
        if (tid < off) red[tid] = fmaxf(red[tid], red[tid + off]);
        __syncthreads();
    }
    const float mx = red[0];
    __syncthreads();
    const float ex0 = __expf(sc[0] - mx);
    const float ex1 = __expf(sc[1] - mx);
    red[tid] = ex0 + ex1;
    __syncthreads();
    for (int off = 128; off > 0; off >>= 1) {
        if (tid < off) red[tid] += red[tid + off];
        __syncthreads();
    }
    const float inv = 1.0f / red[0];

    __bf16* dst = adpT + (size_t)hop * VV * VV;
    dst[(size_t)wids[0] * VV + v] = (__bf16)(ex0 * inv);
    dst[(size_t)wids[1] * VV + v] = (__bf16)(ex1 * inv);
}

// ---------------------------------------------------------------------------
// Kernel 2: one graph-conv hop via bf16 WMMA + double-buffered LDS B panel.
//   Y[slice][w, l] = sum_v adpT[w, v] * Xin[slice][v, l - SHIFT]  (0 if l<SHIFT)
// One block per (n,c) slice, 1024 threads = 32 waves, wave -> one 16-row
// w-tile x full L=64 (4 accumulators). Software pipeline: while computing
// panel k (2x global b128 A + 8x ds_load_b128 B + 4 WMMAs per wave), the
// block cooperatively stages panel k+1 (coalesced, shift applied, stored
// transposed [l][v] with +8 pad) into the other LDS buffer. One barrier per
// K-step covers both the producer->consumer and consumer->overwrite hazards.
// ---------------------------------------------------------------------------
template <bool FIRST, int SHIFT>
__global__ __launch_bounds__(1024) void hop_kernel(
    const float*  __restrict__ xin_f32,   // used when FIRST (original x)
    const __bf16* __restrict__ xin_bf16,  // used when !FIRST (previous hop)
    const __bf16* __restrict__ adpT,      // [V, V] for this hop
    __bf16* __restrict__ yout)            // [B*CIN, V, L]
{
    constexpr int BSTR = 32 + 8;                 // padded row stride (80 B)
    __shared__ __bf16 bt[2][LL][BSTR];           // double-buffered panel

    const int slice = blockIdx.x;                // (n,c) in [0, B*CIN)
    const int tid   = threadIdx.x;
    const int wave  = tid >> 5;                  // 0..31 -> w-tile
    const int lane  = tid & 31;
    const int w0    = wave * 16;

    // WMMA bf16 operand lane mapping (wave32):
    //   A (16x32): m = lane&15, K elems: (lane>>4)*8 +{0..7}, +16 for hi half
    //   B (32x16): n = lane&15, K elems: (lane>>4)*8 +{0..7}, +16 for hi half
    const int mA     = lane & 15;
    const int khalfA = (lane >> 4) * 8;
    const int nB     = lane & 15;
    const int kbB    = (lane >> 4) * 8;

    const size_t  sbase = (size_t)slice * VV * LL;
    const __bf16* arow  = adpT + (size_t)(w0 + mA) * VV + khalfA;

    // staging: thread -> (v = k0 + wave, l = lane*2 .. +1), 256 B / wave
    const int vloc  = wave;
    const int lpair = lane * 2;

    auto stage = [&](int k0, int buf) {
        const size_t rowo = sbase + (size_t)(k0 + vloc) * LL;
        #pragma unroll
        for (int j = 0; j < 2; ++j) {
            const int l  = lpair + j;
            const int ls = l - SHIFT;
            float val = 0.0f;
            if (SHIFT == 0 || ls >= 0) {
                if (FIRST) val = xin_f32[rowo + ls];
                else       val = (float)xin_bf16[rowo + ls];
            }
            bt[buf][l][vloc] = (__bf16)val;
        }
    };

    v8f acc0 = {}, acc1 = {}, acc2 = {}, acc3 = {};

    // prologue: fill buffer 0 with panel k0=0
    stage(0, 0);
    __syncthreads();

    int buf = 0;
    for (int k0 = 0; k0 < VV; k0 += 32, buf ^= 1) {
        // ---- stage next panel into the other buffer (overlaps compute) ----
        if (k0 + 32 < VV) stage(k0 + 32, buf ^ 1);

        // ---- A fragment (2x global_load_b128, prefetch farther ahead) ----
        __builtin_prefetch(arow + k0 + 96, 0, 1);
        v8bf alo = *(const v8bf*)(arow + k0);
        v8bf ahi = *(const v8bf*)(arow + k0 + 16);
        v16bf a;
        #pragma unroll
        for (int i = 0; i < 8; ++i) { a[i] = alo[i]; a[8 + i] = ahi[i]; }

        // ---- 4 output tiles: B via 2x ds_load_b128 each ----
        #pragma unroll
        for (int t = 0; t < 4; ++t) {
            const __bf16* bp = &bt[buf][t * 16 + nB][kbB];
            v8bf blo = *(const v8bf*)(bp);
            v8bf bhi = *(const v8bf*)(bp + 16);
            v16bf b;
            #pragma unroll
            for (int i = 0; i < 8; ++i) { b[i] = blo[i]; b[8 + i] = bhi[i]; }

            v8f* acc = (t == 0) ? &acc0 : (t == 1) ? &acc1 : (t == 2) ? &acc2 : &acc3;
            *acc = __builtin_amdgcn_wmma_f32_16x16x32_bf16(
                       false, a, false, b, (short)0, *acc, false, false);
        }

        // one barrier per step: next panel staged AND current panel consumed
        __syncthreads();
    }

    // C/D layout: VGPR r -> m = r + 8*(lane>>4); n = lane&15
    const int mh = (lane >> 4) * 8;
    const int nn = lane & 15;
    #pragma unroll
    for (int r = 0; r < 8; ++r) {
        const size_t rowo = sbase + (size_t)(w0 + mh + r) * LL;
        yout[rowo +  0 + nn] = (__bf16)acc0[r];
        yout[rowo + 16 + nn] = (__bf16)acc1[r];
        yout[rowo + 32 + nn] = (__bf16)acc2[r];
        yout[rowo + 48 + nn] = (__bf16)acc3[r];
    }
}

// ---------------------------------------------------------------------------
// Kernel 3: 1x1 conv over 128 concatenated channels [x | y0 | y1 | y2].
//   out[n,o,v,l] = mlp_b[o] + sum_c mlp_w[o,c] * h[n,c,v,l]
// One block per (n,v) slice, 512 threads = 16 waves = 16 output tiles.
// Both the weight panel (64x128) and h panel (128x64) are staged in padded
// LDS (bf16), so A and B fragments are all vectorized ds_load_b128.
// ---------------------------------------------------------------------------
__global__ __launch_bounds__(512) void mlp_kernel(
    const float*  __restrict__ x,
    const __bf16* __restrict__ y0,
    const __bf16* __restrict__ y1,
    const __bf16* __restrict__ y2,
    const float*  __restrict__ mlp_w,   // [COUT, CCAT]
    const float*  __restrict__ mlp_b,   // [COUT]
    float* __restrict__ out)            // [B, COUT, V, L]
{
    constexpr int CSTR = CCAT + 8;               // padded row stride (136)
    __shared__ __bf16 hp[LL][CSTR];              // h panel, [l][c]
    __shared__ __bf16 wp[COUT][CSTR];            // weights, [o][c]

    const int slice = blockIdx.x;                // n*V + v
    const int nIdx  = slice >> 9;
    const int v     = slice & (VV - 1);
    const int tid   = threadIdx.x;
    const int wave  = tid >> 5;
    const int lane  = tid & 31;

    // ---- stage weights: thread -> (o = tid>>3, 16 contiguous c) ----
    {
        const int o  = tid >> 3;
        const int c0 = (tid & 7) * 16;
        const float* wr = mlp_w + (size_t)o * CCAT + c0;
        #pragma unroll
        for (int j = 0; j < 16; ++j) wp[o][c0 + j] = (__bf16)wr[j];
    }
    // ---- stage h panel: thread -> (c = tid>>2, 16 contiguous l) ----
    {
        const int c   = tid >> 2;                // 0..127
        const int l0s = (tid & 3) * 16;
        const int g   = c >> 5;                  // concat group (wave-uniform)
        const int cc  = c & 31;
        const size_t off = (((size_t)nIdx * CIN + cc) * VV + v) * LL + l0s;
        #pragma unroll
        for (int j = 0; j < 16; ++j) {
            float val;
            if      (g == 0) val = x[off + j];
            else if (g == 1) val = (float)y0[off + j];
            else if (g == 2) val = (float)y1[off + j];
            else             val = (float)y2[off + j];
            hp[l0s + j][c] = (__bf16)val;
        }
    }
    __syncthreads();

    const int o0 = (wave >> 2) * 16;
    const int l0 = (wave & 3) * 16;

    const int mA     = lane & 15;
    const int khalfA = (lane >> 4) * 8;
    const int nB     = lane & 15;
    const int kbB    = (lane >> 4) * 8;

    v8f acc = {};
    #pragma unroll
    for (int g = 0; g < 4; ++g) {                // K groups of 32 channels
        const int k0 = g * 32;

        const __bf16* ap = &wp[o0 + mA][k0 + khalfA];
        v8bf alo = *(const v8bf*)(ap);
        v8bf ahi = *(const v8bf*)(ap + 16);
        v16bf a;
        #pragma unroll
        for (int i = 0; i < 8; ++i) { a[i] = alo[i]; a[8 + i] = ahi[i]; }

        const __bf16* bp = &hp[l0 + nB][k0 + kbB];
        v8bf blo = *(const v8bf*)(bp);
        v8bf bhi = *(const v8bf*)(bp + 16);
        v16bf b;
        #pragma unroll
        for (int i = 0; i < 8; ++i) { b[i] = blo[i]; b[8 + i] = bhi[i]; }

        acc = __builtin_amdgcn_wmma_f32_16x16x32_bf16(
                  false, a, false, b, (short)0, acc, false, false);
    }

    const int mh = (lane >> 4) * 8;
    const int nn = lane & 15;
    #pragma unroll
    for (int r = 0; r < 8; ++r) {
        const int o = o0 + mh + r;
        out[(((size_t)nIdx * COUT + o) * VV + v) * LL + l0 + nn] = acc[r] + mlp_b[o];
    }
}

// ---------------------------------------------------------------------------
extern "C" void kernel_launch(void* const* d_in, const int* in_sizes, int n_in,
                              void* d_out, int out_size, void* d_ws, size_t ws_size,
                              hipStream_t stream) {
    (void)in_sizes; (void)n_in; (void)out_size; (void)ws_size;

    const float* x        = (const float*)d_in[0];
    const float* nodevec1 = (const float*)d_in[1];
    const float* nodevec2 = (const float*)d_in[2];
    const float* w_trans  = (const float*)d_in[3];
    const float* b_trans  = (const float*)d_in[4];
    const float* mlp_w    = (const float*)d_in[5];
    const float* mlp_b    = (const float*)d_in[6];
    float*       out      = (float*)d_out;

    // workspace layout (bf16): adpT[3][V][V] | y0 | y1 | y2  (~203 MB)
    const size_t NADJ = (size_t)VV * VV;              // 262144
    const size_t NY   = (size_t)BB * CIN * VV * LL;   // 33,554,432
    __bf16* adpT = (__bf16*)d_ws;
    __bf16* y0b  = adpT + 3 * NADJ;
    __bf16* y1b  = y0b + NY;
    __bf16* y2b  = y1b + NY;

    // 1) adjacency (3 hops x 512 rows)
    adj_kernel<<<dim3(3 * VV), 256, 0, stream>>>(nodevec1, nodevec2,
                                                 w_trans, b_trans, adpT);

    // 2-4) graph-conv hops (sequential on stream), 1 block per (n,c) slice
    hop_kernel<true, 0><<<dim3(BB * CIN), 1024, 0, stream>>>(
        x, (const __bf16*)nullptr, adpT + 0 * NADJ, y0b);
    hop_kernel<false, 1><<<dim3(BB * CIN), 1024, 0, stream>>>(
        (const float*)nullptr, y0b, adpT + 1 * NADJ, y1b);
    hop_kernel<false, 2><<<dim3(BB * CIN), 1024, 0, stream>>>(
        (const float*)nullptr, y1b, adpT + 2 * NADJ, y2b);

    // 5) fused concat + 1x1 conv + bias, 1 block per (n,v) slice
    mlp_kernel<<<dim3(BB * VV), 512, 0, stream>>>(
        x, y0b, y1b, y2b, mlp_w, mlp_b, out);
}
